// EdgeWeightPredictor_60129542956
// MI455X (gfx1250) — matrix operand order repaired and checked by reading, not compiled, not run on hardware
//
#include <hip/hip_runtime.h>

#define NN 20000
#define NE 640000
#define EPSV 1e-5f

typedef __attribute__((ext_vector_type(16))) __bf16 v16bf;
typedef __attribute__((ext_vector_type(8)))  float  v8f;

#define AS_LD 132                     // padded f32 LDS stride (kills 16-way bank conflicts)
#define BFRAG_ELEMS (4*8*32*16)       // kc(4) x ntile(8) x lane(32) x half(16) = 16384
#define GEMM_LDS_BYTES (2*BFRAG_ELEMS*2 + 64*AS_LD*4)   // Bhi+Blo (64KB) + A stage (33KB)

static __device__ inline v8f wmma_bf16(v16bf a, v16bf b, v8f c) {
  // (neg_a, A, neg_b, B, c_mod, C, reuse_a, reuse_b)
  return __builtin_amdgcn_wmma_f32_16x16x32_bf16(false, a, false, b, (short)0, c, false, false);
}

// ---- pre-swizzle B = W[row_off + k][n] (f32, ld=128) into WMMA fragment order, hi/lo split
static __device__ inline void fill_bfrag(const float* __restrict__ W, int row_off,
                                         __bf16* Bh, __bf16* Bl, int tid, int nthreads) {
  for (int idx = tid; idx < BFRAG_ELEMS; idx += nthreads) {
    int j  = idx & 15;
    int ln = (idx >> 4) & 31;
    int nt = (idx >> 9) & 7;
    int kc = idx >> 12;
    int k = kc * 32 + (ln >> 4) * 16 + j;     // B 32x16: lanes 0-15 K=0..15, lanes 16-31 K=16..31
    int n = nt * 16 + (ln & 15);
    float f = W[(size_t)(k + row_off) * 128 + n];
    __bf16 h = (__bf16)f;
    Bh[idx] = h;
    Bl[idx] = (__bf16)(f - (float)h);
  }
}

// ---- stage 64 rows x 128 cols of A (f32) into LDS, zero-padding rows >= M
static __device__ inline void fill_astage(const float* __restrict__ A, int M, int tile0,
                                          float* As, int tid, int nthreads) {
  for (int idx = tid; idx < 64 * 32; idx += nthreads) {
    int r = idx >> 5;
    int c = (idx & 31) << 2;
    int row = tile0 + r;
    float4 v = make_float4(0.f, 0.f, 0.f, 0.f);
    if (row < M) v = *(const float4*)(A + (size_t)row * 128 + c);
    float* d = As + r * AS_LD + c;
    d[0] = v.x; d[1] = v.y; d[2] = v.z; d[3] = v.w;
  }
}

// ---- per-wave 16x128 tile, K=128 in 4 chunks of 32, bf16 hi/lo split (3 WMMA per tile-chunk)
static __device__ inline void gemm_tile_compute(const float* As, const __bf16* Bh, const __bf16* Bl,
                                                int wave, int lane, v8f acc[8]) {
  int rrow  = lane & 15;
  int rhalf = lane >> 4;
  const v16bf* Bhf = (const v16bf*)Bh;
  const v16bf* Blf = (const v16bf*)Bl;
  const float* arow = As + (wave * 16 + rrow) * AS_LD;
#pragma unroll
  for (int kc = 0; kc < 4; ++kc) {
    v16bf ah, al;
    int kb = kc * 32 + rhalf * 16;            // A 16x32: lanes 0-15 K=0..15, lanes 16-31 K=16..31
#pragma unroll
    for (int j = 0; j < 16; ++j) {
      float f = arow[kb + j];
      __bf16 h = (__bf16)f;
      ah[j] = h;
      al[j] = (__bf16)(f - (float)h);
    }
#pragma unroll
    for (int nt = 0; nt < 8; ++nt) {
      v16bf bh = Bhf[(kc * 8 + nt) * 32 + lane];
      v16bf bl = Blf[(kc * 8 + nt) * 32 + lane];
      acc[nt] = wmma_bf16(ah, bh, acc[nt]);
      acc[nt] = wmma_bf16(ah, bl, acc[nt]);
      acc[nt] = wmma_bf16(al, bh, acc[nt]);
    }
  }
}

// ================= node-level GEMM: C[M,128] = A[M,128] @ W[row_off:row_off+128, :128]
__global__ __launch_bounds__(128) void k_gemm_node(const float* __restrict__ A, int M,
                                                   const float* __restrict__ W, int row_off,
                                                   float* __restrict__ C) {
  extern __shared__ __align__(32) char smem[];
  __bf16* Bh = (__bf16*)smem;
  __bf16* Bl = Bh + BFRAG_ELEMS;
  float*  As = (float*)(Bl + BFRAG_ELEMS);
  int tid = threadIdx.x;
  fill_bfrag(W, row_off, Bh, Bl, tid, 128);
  int tile0 = blockIdx.x * 64;
  fill_astage(A, M, tile0, As, tid, 128);
  __syncthreads();

  int wave = tid >> 5, lane = tid & 31;
  v8f zero = {0.f, 0.f, 0.f, 0.f, 0.f, 0.f, 0.f, 0.f};
  v8f acc[8];
#pragma unroll
  for (int nt = 0; nt < 8; ++nt) acc[nt] = zero;
  gemm_tile_compute(As, Bh, Bl, wave, lane, acc);

  int rhalf = lane >> 4, rcol = lane & 15;
#pragma unroll
  for (int nt = 0; nt < 8; ++nt)
#pragma unroll
    for (int r = 0; r < 8; ++r) {
      int row = tile0 + wave * 16 + rhalf * 8 + r;   // C layout: VGPR r -> M = r + 8*rhalf
      if (row < M) C[(size_t)row * 128 + nt * 16 + rcol] = acc[nt][r];
    }
}

// ========== fused: u = g@W2 + b2 ; LayerNorm ; ReLU ; out = u@W3 + b3  (E rows, exact 64-tiles)
__global__ __launch_bounds__(128) void k_gemm2_fused(const float* __restrict__ G,
    const float* __restrict__ W2, const float* __restrict__ b2,
    const float* __restrict__ ln_g, const float* __restrict__ ln_b,
    const float* __restrict__ W3, const float* __restrict__ b3,
    float* __restrict__ out) {
  extern __shared__ __align__(32) char smem[];
  __bf16* Bh = (__bf16*)smem;
  __bf16* Bl = Bh + BFRAG_ELEMS;
  float*  As = (float*)(Bl + BFRAG_ELEMS);
  int tid = threadIdx.x;
  fill_bfrag(W2, 0, Bh, Bl, tid, 128);
  int tile0 = blockIdx.x * 64;
  fill_astage(G, NE, tile0, As, tid, 128);
  __syncthreads();

  int wave = tid >> 5, lane = tid & 31;
  v8f zero = {0.f, 0.f, 0.f, 0.f, 0.f, 0.f, 0.f, 0.f};
  v8f acc[8];
#pragma unroll
  for (int nt = 0; nt < 8; ++nt) acc[nt] = zero;
  gemm_tile_compute(As, Bh, Bl, wave, lane, acc);
  __syncthreads();

  // dump accumulators (+b2) back to LDS (reuse A-stage region) in row-major for the LN epilogue
  int rhalf = lane >> 4, rcol = lane & 15;
  float* U = As;
#pragma unroll
  for (int nt = 0; nt < 8; ++nt) {
    int col = nt * 16 + rcol;
    float bias = b2[col];
#pragma unroll
    for (int r = 0; r < 8; ++r)
      U[(wave * 16 + rhalf * 8 + r) * AS_LD + col] = acc[nt][r] + bias;
  }
  // each wave reads only its own 16 rows (written by itself) -> no barrier needed
  float b3v = b3[0];
  for (int r = 0; r < 16; ++r) {
    const float* u = U + (wave * 16 + r) * AS_LD;
    float s = 0.f, sq = 0.f;
#pragma unroll
    for (int c = lane; c < 128; c += 32) { float v = u[c]; s += v; sq += v * v; }
#pragma unroll
    for (int off = 16; off > 0; off >>= 1) { s += __shfl_xor(s, off); sq += __shfl_xor(sq, off); }
    float mu  = s  * (1.0f / 128.0f);
    float var = sq * (1.0f / 128.0f) - mu * mu;
    float rs  = rsqrtf(var + EPSV);
    float dot = 0.f;
#pragma unroll
    for (int c = lane; c < 128; c += 32) {
      float w = (u[c] - mu) * rs * ln_g[c] + ln_b[c];
      w = fmaxf(w, 0.0f);
      dot += w * W3[c];
    }
#pragma unroll
    for (int off = 16; off > 0; off >>= 1) dot += __shfl_xor(dot, off);
    if (lane == 0) out[tile0 + wave * 16 + r] = dot + b3v;
  }
}

// ================= scalar / scatter kernels =================
__global__ void k_zero(float* p, int n) {
  int i = blockIdx.x * blockDim.x + threadIdx.x;
  if (i < n) p[i] = 0.0f;
}

__global__ void k_deg(const long long* __restrict__ ei, float* __restrict__ deg) {
  int e = blockIdx.x * blockDim.x + threadIdx.x;
  if (e < NE) atomicAdd(deg + (int)ei[NE + e], 1.0f);
}

__global__ void k_dinv(float* deg) {
  int i = blockIdx.x * blockDim.x + threadIdx.x;
  if (i < NN) deg[i] = rsqrtf(deg[i] + 1.0f);
}

// h1[e] = (y1[src_e] + y2[dst_e]) * selfnorm_e + b1   (full overwrite; 32 lanes x float4 per edge)
__global__ void k_conv1_self(const long long* __restrict__ ei, const float* __restrict__ y1,
                             const float* __restrict__ y2, const float* __restrict__ dinv,
                             const float* __restrict__ b1, float* __restrict__ h1) {
  long long t = (long long)blockIdx.x * blockDim.x + threadIdx.x;
  int e = (int)(t >> 5);
  if (e >= NE) return;
  int q = ((int)t & 31) << 2;
  int s = (int)ei[e], d = (int)ei[NE + e];
  float se = (e < NN) ? dinv[e] * dinv[e] : 1.0f;
  float4 a  = *(const float4*)(y1 + (size_t)s * 128 + q);
  float4 b  = *(const float4*)(y2 + (size_t)d * 128 + q);
  float4 bb = *(const float4*)(b1 + q);
  float4 o = make_float4((a.x + b.x) * se + bb.x, (a.y + b.y) * se + bb.y,
                         (a.z + b.z) * se + bb.z, (a.w + b.w) * se + bb.w);
  *(float4*)(h1 + (size_t)e * 128 + q) = o;
}

// h1[dst_e] += m[src_e] * norm_e  with m[j] = y1[src_j] + y2[dst_j]  (src_e < NN <= NE)
__global__ void k_conv1_scatter(const long long* __restrict__ ei, const float* __restrict__ y1,
                                const float* __restrict__ y2, const float* __restrict__ dinv,
                                float* __restrict__ h1) {
  long long t = (long long)blockIdx.x * blockDim.x + threadIdx.x;
  int e = (int)(t >> 5);
  if (e >= NE) return;
  int q = ((int)t & 31) << 2;
  int s = (int)ei[e], d = (int)ei[NE + e];
  float nrm = dinv[s] * dinv[d];
  int s2 = (int)ei[s], d2 = (int)ei[NE + s];
  float4 a = *(const float4*)(y1 + (size_t)s2 * 128 + q);
  float4 b = *(const float4*)(y2 + (size_t)d2 * 128 + q);
  float* dst = h1 + (size_t)d * 128 + q;
  atomicAdd(dst + 0, (a.x + b.x) * nrm);
  atomicAdd(dst + 1, (a.y + b.y) * nrm);
  atomicAdd(dst + 2, (a.z + b.z) * nrm);
  atomicAdd(dst + 3, (a.w + b.w) * nrm);
}

__global__ void k_bn_reduce(const float* __restrict__ h1, float* __restrict__ sums) {
  int c = threadIdx.x;                        // 128 channels
  float s = 0.f, sq = 0.f;
  for (int r = blockIdx.x; r < NE; r += gridDim.x) {
    float v = h1[(size_t)r * 128 + c];
    s += v; sq += v * v;
  }
  atomicAdd(&sums[c], s);
  atomicAdd(&sums[128 + c], sq);
}

__global__ void k_bn_final(const float* __restrict__ sums, const float* __restrict__ bn_g,
                           const float* __restrict__ bn_b, float* __restrict__ sc_sh) {
  int c = threadIdx.x;
  float mean = sums[c] * (1.0f / NE);
  float var  = sums[128 + c] * (1.0f / NE) - mean * mean;
  float sc = bn_g[c] * rsqrtf(var + EPSV);
  sc_sh[c] = sc;
  sc_sh[128 + c] = bn_b[c] - mean * sc;
}

// snapshot z rows < NN before h1 is overwritten in place
__global__ void k_z_small(const float* __restrict__ h1, const float* __restrict__ sc_sh,
                          float* __restrict__ z) {
  int t = blockIdx.x * blockDim.x + threadIdx.x;
  if (t >= NN * 128) return;
  int c = t & 127;
  z[t] = fmaxf(h1[t] * sc_sh[c] + sc_sh[128 + c], 0.0f);
}

// in-place: h1[e] = relu(BN(h1[e])) * selfnorm_e
__global__ void k_conv2_self(const float* __restrict__ dinv, const float* __restrict__ sc_sh,
                             float* __restrict__ h1) {
  long long t = (long long)blockIdx.x * blockDim.x + threadIdx.x;
  int e = (int)(t >> 5);
  if (e >= NE) return;
  int q = ((int)t & 31) << 2;
  float se = (e < NN) ? dinv[e] * dinv[e] : 1.0f;
  float* p = h1 + (size_t)e * 128 + q;
#pragma unroll
  for (int i = 0; i < 4; ++i) {
    int c = q + i;
    float v = fmaxf(p[i] * sc_sh[c] + sc_sh[128 + c], 0.0f);
    p[i] = v * se;
  }
}

// h1[dst_e] += z[src_e] * norm_e
__global__ void k_conv2_scatter(const long long* __restrict__ ei, const float* __restrict__ dinv,
                                const float* __restrict__ z, float* __restrict__ h1) {
  long long t = (long long)blockIdx.x * blockDim.x + threadIdx.x;
  int e = (int)(t >> 5);
  if (e >= NE) return;
  int q = ((int)t & 31) << 2;
  int s = (int)ei[e], d = (int)ei[NE + e];
  float nrm = dinv[s] * dinv[d];
  float4 zz = *(const float4*)(z + (size_t)s * 128 + q);
  float* dst = h1 + (size_t)d * 128 + q;
  atomicAdd(dst + 0, zz.x * nrm);
  atomicAdd(dst + 1, zz.y * nrm);
  atomicAdd(dst + 2, zz.z * nrm);
  atomicAdd(dst + 3, zz.w * nrm);
}

extern "C" void kernel_launch(void* const* d_in, const int* in_sizes, int n_in,
                              void* d_out, int out_size, void* d_ws, size_t ws_size,
                              hipStream_t stream) {
  (void)in_sizes; (void)n_in; (void)out_size; (void)ws_size;
  const float*     x    = (const float*)d_in[0];
  const long long* ei   = (const long long*)d_in[1];
  const float*     W1   = (const float*)d_in[2];
  const float*     b1   = (const float*)d_in[3];
  const float*     bn_g = (const float*)d_in[4];
  const float*     bn_b = (const float*)d_in[5];
  const float*     W2   = (const float*)d_in[6];
  const float*     b2   = (const float*)d_in[7];
  const float*     ln_g = (const float*)d_in[8];
  const float*     ln_b = (const float*)d_in[9];
  const float*     W3   = (const float*)d_in[10];
  const float*     b3   = (const float*)d_in[11];
  float* out = (float*)d_out;

  float* ws    = (float*)d_ws;
  float* dinv  = ws;                          // 20480 floats (20000 used)
  float* sums  = ws + 20480;                  // 256 (bn sum | sumsq)
  float* sc_sh = sums + 256;                  // 256 (bn scale | shift)
  float* y1    = sc_sh + 256;                 // NN*128
  float* y2    = y1 + (size_t)NN * 128;       // NN*128
  float* zs    = y2 + (size_t)NN * 128;       // NN*128
  float* h1    = zs + (size_t)NN * 128;       // NE*128  (reused in place as g)

  int blkE = (int)(((long long)NE * 32 + 255) / 256);   // 80000 blocks of 256

  k_zero<<<(20992 + 255) / 256, 256, 0, stream>>>(ws, 20992);
  k_deg<<<(NE + 255) / 256, 256, 0, stream>>>(ei, dinv);
  k_dinv<<<(NN + 255) / 256, 256, 0, stream>>>(dinv);

  k_gemm_node<<<(NN + 63) / 64, 128, GEMM_LDS_BYTES, stream>>>(x, NN, W1, 0,   y1);
  k_gemm_node<<<(NN + 63) / 64, 128, GEMM_LDS_BYTES, stream>>>(x, NN, W1, 128, y2);

  k_conv1_self<<<blkE, 256, 0, stream>>>(ei, y1, y2, dinv, b1, h1);
  k_conv1_scatter<<<blkE, 256, 0, stream>>>(ei, y1, y2, dinv, h1);

  k_bn_reduce<<<1024, 128, 0, stream>>>(h1, sums);
  k_bn_final<<<1, 128, 0, stream>>>(sums, bn_g, bn_b, sc_sh);

  k_z_small<<<(NN * 128 + 255) / 256, 256, 0, stream>>>(h1, sc_sh, zs);
  k_conv2_self<<<blkE, 256, 0, stream>>>(dinv, sc_sh, h1);
  k_conv2_scatter<<<blkE, 256, 0, stream>>>(ei, dinv, zs, h1);

  k_gemm2_fused<<<NE / 64, 128, GEMM_LDS_BYTES, stream>>>(h1, W2, b2, ln_g, ln_b, W3, b3, out);
}
